// GPT4TSModel_41180146434222
// MI455X (gfx1250) — matrix operand order repaired and checked by reading, not compiled, not run on hardware
//
#include <hip/hip_runtime.h>
#include <math.h>

// ---------------------------------------------------------------------------
// GPT4TS encoder forward for MI455X (gfx1250, wave32).
// GEMMs: f16 operands, v_wmma_f32_16x16x32_f16, LDS tiles double-buffered and
// filled by the Tensor Data Mover (tensor_load_to_lds + s_wait_tensorcnt).
// Weights converted fp32->f16 once per launch (L2-resident, ~85MB of f16).
// ---------------------------------------------------------------------------

typedef __attribute__((ext_vector_type(16))) _Float16 v16h;
typedef __attribute__((ext_vector_type(8)))  float    v8f;
typedef unsigned int u32x4 __attribute__((ext_vector_type(4)));
typedef int          i32x8 __attribute__((ext_vector_type(8)));
typedef int          i32x4 __attribute__((ext_vector_type(4)));

#define D_EMB   768
#define NPATCH  16
#define TLEN    8192
#define NTOK    512
#define NHEAD   12
#define HDIM    64
#define NLAYER  6
#define DFF_    3072
#define BATCH   8
#define MROWS   (BATCH * NTOK)      // 4096 rows
#define LN_EPS  1e-5f

#define LDT 40   // LDS tile pitch in halves: 32 data + 4-DWORD TDM pad = 80B

// ---- WMMA fragment helpers (ISA 7.12.2, wave32) ----------------------------
// A (16x32): lanes 0-15 -> row=lane, Ks {0..15}; lanes 16-31 Ks {16..31}
__device__ __forceinline__ v16h frag_a(const _Float16* T, int row0, int lane) {
    const int half = lane >> 4, r = lane & 15;
    const _Float16* p = T + (row0 + r) * LDT;
    v16h a;
#pragma unroll
    for (int i = 0; i < 8; ++i) {
        a[i]     = p[half * 16 + i];
        a[8 + i] = p[half * 16 + 8 + i];
    }
    return a;
}
// B (32x16) from row-major W-tile [n][k]: lane col n=lane&15, k=half*16+i (contig)
__device__ __forceinline__ v16h frag_b(const _Float16* T, int col0, int lane) {
    const int half = lane >> 4, n = lane & 15;
    const _Float16* p = T + (col0 + n) * LDT + half * 16;
    v16h b;
#pragma unroll
    for (int i = 0; i < 16; ++i) b[i] = p[i];
    return b;
}

// ---- TDM: DMA one 128x32 f16 tile (row-major, K-pitch) into LDS ------------
// D# group0/group1 packed per ISA 8.3/8.4; pad 4 DWORDs per 16-DWORD row so
// the LDS image lands at an 80B pitch (LDT). Issued by wave 0 only.
__device__ __forceinline__ void tdm_tile_load(const _Float16* gbase, int row0,
                                              int K, int rowsTot, unsigned lds_off,
                                              int kt) {
    const unsigned long long ga =
        (unsigned long long)(uintptr_t)(gbase + (size_t)row0 * K + kt);
    u32x4 g0;
    g0[0] = 1u;                                     // count=1, user descriptor
    g0[1] = lds_off;                                // LDS byte address
    g0[2] = (unsigned)ga;                           // global_addr[31:0]
    g0[3] = (unsigned)((ga >> 32) & 0x1FFFFFFu) | (2u << 30);  // addr[56:32] | type=2
    i32x8 g1;
    g1[0] = (1 << 16)      // data_size = 2B
          | (1 << 20)      // pad_enable
          | (3 << 22)      // pad_interval: 16 DWORDs (one 64B row)
          | (3 << 25);     // pad_amount: 4 DWORDs (16B)
    g1[1] = (K & 0xffff) << 16;                     // tensor_dim0 lo16
    g1[2] = ((unsigned)K >> 16) | ((rowsTot & 0xffff) << 16);  // dim0 hi | dim1 lo
    g1[3] = ((unsigned)rowsTot >> 16) | (32 << 16); // dim1 hi | tile_dim0=32
    g1[4] = 128;                                    // tile_dim1=128, tile_dim2=0
    g1[5] = K;                                      // tensor_dim0_stride lo32
    g1[6] = 0;
    g1[7] = 0;
    const i32x4 z4 = {0, 0, 0, 0};
    const i32x8 z8 = {0, 0, 0, 0, 0, 0, 0, 0};
    __builtin_amdgcn_tensor_load_to_lds(g0, g1, z4, z4, z8, 0);
}

// ---- Tiled WMMA GEMM:  C[M,N] = epi( A[M,K](f16) * W[N,K]^T(f16) (+Res) ) --
// 256 thr = 8 waves; block tile 128x128; wave tile 64x32 (4x2 wmma accums).
// TDM double-buffers the A/W tiles; wave0 drives the DMA + TENSORcnt waits.
// EPI: 0 plain, 1 +Res(fp32), 2 exact-gelu.  OutT: _Float16 or float.
template <int EPI, typename OutT>
__global__ __launch_bounds__(256) void gemm_wmma(
    const _Float16* __restrict__ A, const _Float16* __restrict__ W,
    const float* __restrict__ Res, OutT* __restrict__ C,
    int M, int N, int K)
{
    __shared__ _Float16 smemA[2][128 * LDT];
    __shared__ _Float16 smemW[2][128 * LDT];

    const int tid  = threadIdx.x;
    const int lane = tid & 31;
    const int wave = tid >> 5;
    const int wm   = (wave >> 2) * 64;   // 0 / 64
    const int wn   = (wave & 3) * 32;    // 0..96
    const int rowB = blockIdx.y * 128;
    const int colB = blockIdx.x * 128;

    v8f acc[4][2];
#pragma unroll
    for (int mt = 0; mt < 4; ++mt)
#pragma unroll
        for (int nt = 0; nt < 2; ++nt) acc[mt][nt] = v8f{0, 0, 0, 0, 0, 0, 0, 0};

    const int nk = K >> 5;
    if (wave == 0) {  // prologue: DMA tile 0 into buffer 0
        tdm_tile_load(A, rowB, K, M, (unsigned)(uintptr_t)&smemA[0][0], 0);
        tdm_tile_load(W, colB, K, N, (unsigned)(uintptr_t)&smemW[0][0], 0);
    }

    for (int it = 0; it < nk; ++it) {
        const int buf = it & 1;
        if (wave == 0) __builtin_amdgcn_s_wait_tensorcnt(0);  // tile `it` landed
        __syncthreads();
        if (wave == 0 && it + 1 < nk) {  // overlap: DMA tile it+1 into other buf
            tdm_tile_load(A, rowB, K, M,
                          (unsigned)(uintptr_t)&smemA[buf ^ 1][0], (it + 1) * 32);
            tdm_tile_load(W, colB, K, N,
                          (unsigned)(uintptr_t)&smemW[buf ^ 1][0], (it + 1) * 32);
        }
        const _Float16* As = &smemA[buf][0];
        const _Float16* Ws = &smemW[buf][0];
        const v16h b0 = frag_b(Ws, wn,      lane);
        const v16h b1 = frag_b(Ws, wn + 16, lane);
#pragma unroll
        for (int mt = 0; mt < 4; ++mt) {
            const v16h a = frag_a(As, wm + mt * 16, lane);
            acc[mt][0] = __builtin_amdgcn_wmma_f32_16x16x32_f16(false, a, false, b0,
                                                                (short)0, acc[mt][0], false, false);
            acc[mt][1] = __builtin_amdgcn_wmma_f32_16x16x32_f16(false, a, false, b1,
                                                                (short)0, acc[mt][1], false, false);
        }
        __syncthreads();  // all waves done reading `buf` before TDM reuses it
    }

    // epilogue + store (C/D layout: row = half*8 + j, col = lane&15)
    const int half = lane >> 4, n = lane & 15;
#pragma unroll
    for (int mt = 0; mt < 4; ++mt)
#pragma unroll
        for (int nt = 0; nt < 2; ++nt) {
            const int col = colB + wn + nt * 16 + n;
#pragma unroll
            for (int j = 0; j < 8; ++j) {
                const int row = rowB + wm + mt * 16 + half * 8 + j;
                float v = acc[mt][nt][j];
                if (EPI == 1) v += Res[(size_t)row * N + col];
                if (EPI == 2) v = 0.5f * v * (1.0f + erff(v * 0.70710678118654752f));
                C[(size_t)row * N + col] = (OutT)v;
            }
        }
}

// ---- Causal attention: one wave per (b, h, 16-query tile), f16 in/out ------
__global__ __launch_bounds__(32) void attn_kernel(
    const _Float16* __restrict__ qkv, _Float16* __restrict__ out)
{
    __shared__ float S[16 * (NTOK + 4)];
    constexpr int SLD = NTOK + 4;

    const int lane = threadIdx.x;
    int idx = blockIdx.x;
    const int qt = idx & 31; idx >>= 5;          // NTOK/16 query tiles
    const int h  = idx % NHEAD;
    const int b  = idx / NHEAD;
    const int qbase = qt * 16;
    const int kend  = qbase + 16;
    const int kend2 = (kend + 31) & ~31;

    const size_t RS = 3 * D_EMB;
    const _Float16* Qb = qkv + (size_t)b * NTOK * RS + (size_t)h * HDIM;
    const _Float16* Kb = Qb + D_EMB;
    const _Float16* Vb = Qb + 2 * D_EMB;
    const int half = lane >> 4, l15 = lane & 15;

    v16h aq[2];
#pragma unroll
    for (int c = 0; c < 2; ++c) {
        const _Float16* p = Qb + (size_t)(qbase + l15) * RS + c * 32;
#pragma unroll
        for (int i = 0; i < 8; ++i) {
            aq[c][i]     = (_Float16)((float)p[half * 16 + i] * 0.125f);
            aq[c][8 + i] = (_Float16)((float)p[half * 16 + 8 + i] * 0.125f);
        }
    }

    for (int ct = 0; ct < kend / 16; ++ct) {
        v16h bk0, bk1;
        const _Float16* kp = Kb + (size_t)(ct * 16 + l15) * RS;
#pragma unroll
        for (int i = 0; i < 16; ++i) {
            bk0[i] = kp[half * 16 + i];
            bk1[i] = kp[32 + half * 16 + i];
        }
        v8f s = v8f{0, 0, 0, 0, 0, 0, 0, 0};
        s = __builtin_amdgcn_wmma_f32_16x16x32_f16(false, aq[0], false, bk0, (short)0, s, false, false);
        s = __builtin_amdgcn_wmma_f32_16x16x32_f16(false, aq[1], false, bk1, (short)0, s, false, false);
#pragma unroll
        for (int j = 0; j < 8; ++j) {
            const int row = half * 8 + j;
            const int col = ct * 16 + l15;
            S[row * SLD + col] = (col <= qbase + row) ? s[j] : -1e30f;
        }
    }
    for (int c = kend + lane; c < kend2; c += 32)
#pragma unroll
        for (int r = 0; r < 16; ++r) S[r * SLD + c] = 0.0f;
    __syncthreads();

    if (lane < 16) {
        float* row = &S[lane * SLD];
        float m = -1e30f;
        for (int c = 0; c < kend; ++c) m = fmaxf(m, row[c]);
        float sum = 0.0f;
        for (int c = 0; c < kend; ++c) { const float e = expf(row[c] - m); row[c] = e; sum += e; }
        const float inv = 1.0f / sum;
        for (int c = 0; c < kend; ++c) row[c] *= inv;
    }
    __syncthreads();

    v8f o[4];
#pragma unroll
    for (int nt = 0; nt < 4; ++nt) o[nt] = v8f{0, 0, 0, 0, 0, 0, 0, 0};
    for (int kc = 0; kc < kend2; kc += 32) {
        v16h a;
        const float* sp = &S[l15 * SLD + kc];
#pragma unroll
        for (int i = 0; i < 8; ++i) {
            a[i]     = (_Float16)sp[half * 16 + i];
            a[8 + i] = (_Float16)sp[half * 16 + 8 + i];
        }
#pragma unroll
        for (int nt = 0; nt < 4; ++nt) {
            v16h bv;
            const _Float16* vp = Vb + (size_t)(kc + half * 16) * RS + nt * 16 + l15;
#pragma unroll
            for (int i = 0; i < 16; ++i) bv[i] = vp[(size_t)i * RS];
            o[nt] = __builtin_amdgcn_wmma_f32_16x16x32_f16(false, a, false, bv, (short)0, o[nt], false, false);
        }
    }
    _Float16* ob = out + ((size_t)b * NTOK + qbase) * D_EMB + h * HDIM;
#pragma unroll
    for (int nt = 0; nt < 4; ++nt)
#pragma unroll
        for (int j = 0; j < 8; ++j)
            ob[(size_t)(half * 8 + j) * D_EMB + nt * 16 + l15] = (_Float16)o[nt][j];
}

// ---- LayerNorm: fp32 in, f16 out (GEMM operand precision) ------------------
__global__ __launch_bounds__(256) void ln_kernel(
    const float* __restrict__ x, const float* __restrict__ g,
    const float* __restrict__ bsh, _Float16* __restrict__ y)
{
    __shared__ float red[256];
    const int row = blockIdx.x, tid = threadIdx.x;
    const float* xr = x + (size_t)row * D_EMB;

    float s = 0.0f;
    for (int c = tid; c < D_EMB; c += 256) s += xr[c];
    red[tid] = s; __syncthreads();
    for (int off = 128; off > 0; off >>= 1) {
        if (tid < off) red[tid] += red[tid + off];
        __syncthreads();
    }
    const float mean = red[0] / (float)D_EMB;
    __syncthreads();

    float v = 0.0f;
    for (int c = tid; c < D_EMB; c += 256) { const float d = xr[c] - mean; v += d * d; }
    red[tid] = v; __syncthreads();
    for (int off = 128; off > 0; off >>= 1) {
        if (tid < off) red[tid] += red[tid + off];
        __syncthreads();
    }
    const float rstd = rsqrtf(red[0] / (float)D_EMB + LN_EPS);

    for (int c = tid; c < D_EMB; c += 256)
        y[(size_t)row * D_EMB + c] = (_Float16)((xr[c] - mean) * rstd * g[c] + bsh[c]);
}

// ---- fp32 -> f16 conversion (weights; once per launch) ---------------------
__global__ __launch_bounds__(256) void cvt_f16(
    const float* __restrict__ src, _Float16* __restrict__ dst, int n)
{
    const int i = blockIdx.x * 256 + threadIdx.x;
    if (i < n) dst[i] = (_Float16)src[i];
}

// ---- Patch embed (K=16, VALU) ----------------------------------------------
__global__ __launch_bounds__(256) void patch_embed(
    const float* __restrict__ x, const float* __restrict__ pw,
    const float* __restrict__ pb, const float* __restrict__ pe,
    float* __restrict__ z)
{
    const int idx = blockIdx.x * 256 + threadIdx.x;
    if (idx >= MROWS * D_EMB) return;
    const int d = idx % D_EMB, bn = idx / D_EMB;
    const int b = bn / NTOK, n = bn % NTOK;
    const float* xp = x + (size_t)b * TLEN + n * NPATCH;
    const float* wp = pw + (size_t)d * NPATCH;
    float s = pb[d] + pe[(size_t)n * D_EMB + d];
#pragma unroll
    for (int p = 0; p < NPATCH; ++p) s += xp[p] * wp[p];
    z[idx] = s;
}

// ---- Token-mean pooling (f16 in, fp32 out) ---------------------------------
__global__ __launch_bounds__(256) void mean_kernel(
    const _Float16* __restrict__ hf, float* __restrict__ enc)
{
    const int idx = blockIdx.x * 256 + threadIdx.x;
    if (idx >= BATCH * D_EMB) return;
    const int b = idx / D_EMB, d = idx % D_EMB;
    float s = 0.0f;
    for (int n = 0; n < NTOK; ++n) s += (float)hf[((size_t)b * NTOK + n) * D_EMB + d];
    enc[idx] = s * (1.0f / (float)NTOK);
}

// ---- Head MLP --------------------------------------------------------------
__global__ __launch_bounds__(256) void head1_kernel(
    const float* __restrict__ enc, const float* __restrict__ w,
    const float* __restrict__ bias, float* __restrict__ y1)
{
    const int idx = blockIdx.x * 256 + threadIdx.x;
    if (idx >= BATCH * (D_EMB / 2)) return;
    const int b = idx / (D_EMB / 2), j = idx % (D_EMB / 2);
    const float* e = enc + (size_t)b * D_EMB;
    const float* wr = w + (size_t)j * D_EMB;
    float s = bias[j];
    for (int k = 0; k < D_EMB; ++k) s += e[k] * wr[k];
    y1[idx] = fmaxf(s, 0.0f);
}

__global__ __launch_bounds__(32) void head2_kernel(
    const float* __restrict__ y1, const float* __restrict__ w,
    const float* __restrict__ bias, float* __restrict__ out)
{
    const int idx = threadIdx.x;
    if (idx >= BATCH * 2) return;
    const int b = idx / 2, o = idx % 2;
    const float* yr = y1 + (size_t)b * (D_EMB / 2);
    const float* wr = w + (size_t)o * (D_EMB / 2);
    float s = bias[o];
    for (int k = 0; k < D_EMB / 2; ++k) s += yr[k] * wr[k];
    out[idx] = s;
}

// ---------------------------------------------------------------------------
extern "C" void kernel_launch(void* const* d_in, const int* in_sizes, int n_in,
                              void* d_out, int out_size, void* d_ws, size_t ws_size,
                              hipStream_t stream)
{
    const float* x     = (const float*)d_in[0];
    const float* pp_w  = (const float*)d_in[1];
    const float* pp_b  = (const float*)d_in[2];
    const float* pe    = (const float*)d_in[3];
    const float* ln1_g = (const float*)d_in[4];
    const float* ln1_b = (const float*)d_in[5];
    const float* qkv_w = (const float*)d_in[6];
    const float* out_w = (const float*)d_in[7];
    const float* ln2_g = (const float*)d_in[8];
    const float* ln2_b = (const float*)d_in[9];
    const float* fc1_w = (const float*)d_in[10];
    const float* fc2_w = (const float*)d_in[11];
    const float* lnf_g = (const float*)d_in[12];
    const float* lnf_b = (const float*)d_in[13];
    const float* h1_w  = (const float*)d_in[14];
    const float* h1_b  = (const float*)d_in[15];
    const float* h2_w  = (const float*)d_in[16];
    const float* h2_b  = (const float*)d_in[17];

    // workspace carve: z(f32) | hb(f16) | big(f16) | enc | y1 | f16 weights
    float*    z   = (float*)d_ws;
    _Float16* hb  = (_Float16*)(z + (size_t)MROWS * D_EMB);
    _Float16* big = hb + (size_t)MROWS * D_EMB;
    float*    enc = (float*)(big + (size_t)MROWS * DFF_);
    float*    y1  = enc + (size_t)BATCH * D_EMB;
    _Float16* wq  = (_Float16*)(y1 + (size_t)BATCH * (D_EMB / 2));
    _Float16* wo  = wq + (size_t)NLAYER * 3 * D_EMB * D_EMB;
    _Float16* w1  = wo + (size_t)NLAYER * D_EMB * D_EMB;
    _Float16* w2  = w1 + (size_t)NLAYER * DFF_ * D_EMB;

    const dim3 blk256(256);
    // weight conversion (f16 GEMM operands, L2-resident)
    {
        const int nq = NLAYER * 3 * D_EMB * D_EMB;
        const int no = NLAYER * D_EMB * D_EMB;
        const int nf = NLAYER * DFF_ * D_EMB;
        cvt_f16<<<(nq + 255) / 256, blk256, 0, stream>>>(qkv_w, wq, nq);
        cvt_f16<<<(no + 255) / 256, blk256, 0, stream>>>(out_w, wo, no);
        cvt_f16<<<(nf + 255) / 256, blk256, 0, stream>>>(fc1_w, w1, nf);
        cvt_f16<<<(nf + 255) / 256, blk256, 0, stream>>>(fc2_w, w2, nf);
    }

    patch_embed<<<(MROWS * D_EMB + 255) / 256, blk256, 0, stream>>>(x, pp_w, pp_b, pe, z);

    for (int i = 0; i < NLAYER; ++i) {
        const _Float16* Wq = wq + (size_t)i * 3 * D_EMB * D_EMB;
        const _Float16* Wo = wo + (size_t)i * D_EMB * D_EMB;
        const _Float16* W1 = w1 + (size_t)i * DFF_ * D_EMB;
        const _Float16* W2 = w2 + (size_t)i * D_EMB * DFF_;

        ln_kernel<<<MROWS, blk256, 0, stream>>>(z, ln1_g + i * D_EMB, ln1_b + i * D_EMB, hb);
        gemm_wmma<0, _Float16><<<dim3(3 * D_EMB / 128, MROWS / 128), blk256, 0, stream>>>(
            hb, Wq, nullptr, big, MROWS, 3 * D_EMB, D_EMB);
        attn_kernel<<<BATCH * NHEAD * (NTOK / 16), 32, 0, stream>>>(big, hb);
        gemm_wmma<1, float><<<dim3(D_EMB / 128, MROWS / 128), blk256, 0, stream>>>(
            hb, Wo, z, z, MROWS, D_EMB, D_EMB);
        ln_kernel<<<MROWS, blk256, 0, stream>>>(z, ln2_g + i * D_EMB, ln2_b + i * D_EMB, hb);
        gemm_wmma<2, _Float16><<<dim3(DFF_ / 128, MROWS / 128), blk256, 0, stream>>>(
            hb, W1, nullptr, big, MROWS, DFF_, D_EMB);
        gemm_wmma<1, float><<<dim3(D_EMB / 128, MROWS / 128), blk256, 0, stream>>>(
            big, W2, z, z, MROWS, D_EMB, DFF_);
    }

    ln_kernel<<<MROWS, blk256, 0, stream>>>(z, lnf_g, lnf_b, hb);
    mean_kernel<<<(BATCH * D_EMB + 255) / 256, blk256, 0, stream>>>(hb, enc);
    head1_kernel<<<(BATCH * (D_EMB / 2) + 255) / 256, blk256, 0, stream>>>(enc, h1_w, h1_b, y1);
    head2_kernel<<<1, 32, 0, stream>>>(y1, h2_w, h2_b, (float*)d_out);
}
